// CasualSelfAttention_36464272343466
// MI455X (gfx1250) — matrix operand order repaired and checked by reading, not compiled, not run on hardware
//
#include <hip/hip_runtime.h>
#include <hip/hip_bf16.h>

typedef __attribute__((ext_vector_type(16))) _Float16 v16h;
typedef __attribute__((ext_vector_type(8)))  float    v8f;

#define NEMBD 1024
#define NHEAD 16
#define HDIM  64
#define TSEQ  2048
#define BSZ   4

__device__ inline v8f zero8() {
  v8f z;
#pragma unroll
  for (int i = 0; i < 8; ++i) z[i] = 0.0f;
  return z;
}

__device__ inline v8f wmma16(v16h a, v16h b, v8f c) {
  // D = A(16x32 f16) * B(32x16 f16) + C(16x16 f32)
  return __builtin_amdgcn_wmma_f32_16x16x32_f16(
      /*neg_a=*/false, a, /*neg_b=*/false, b,
      /*c_mod=*/(short)0, c, /*reuse_a=*/false, /*reuse_b=*/false);
}

// ---- CDNA5 async global->LDS copies (ASYNCcnt tracked, no VGPR data path) ----
__device__ __forceinline__ void async_copy_b128(unsigned lds_off, const void* gaddr) {
  asm volatile("global_load_async_to_lds_b128 %0, %1, off"
               :: "v"(lds_off), "v"(gaddr) : "memory");
}
// async loads complete in order -> waiting "<= N" guarantees everything older
// than the newest N copies has landed in LDS.
__device__ __forceinline__ void wait_async_le0() {
  asm volatile("s_wait_asynccnt 0" ::: "memory");
}
__device__ __forceinline__ void wait_async_le2() {
  asm volatile("s_wait_asynccnt 2" ::: "memory");
}
__device__ __forceinline__ void wait_async_le4() {
  asm volatile("s_wait_asynccnt 4" ::: "memory");
}
// LDS aperture sits in addr[63:32]; low 32 bits of a generic LDS pointer are
// the LDS byte offset (ISA 10.2 aperture mapping).
__device__ __forceinline__ unsigned lds_off_of(const void* p) {
  return (unsigned)(unsigned long long)p;
}

// A fragment 16x32 f16, row-major source with leading dim ldm.
// lanes 0-15: M=lane, K in {0..7} U {16..23}; lanes 16-31: M=lane-16, K+8.
__device__ __forceinline__ v16h load_afrag(const _Float16* p0, int ldm) {
  const int lane = threadIdx.x & 31;
  const int m  = lane & 15;
  const int kb = (lane >> 4) << 3;  // 0 or 8
  const _Float16* p = p0 + (size_t)m * ldm + kb;
  v16h a;
#pragma unroll
  for (int i = 0; i < 8; ++i) { a[i] = p[i]; a[8 + i] = p[16 + i]; }
  return a;
}

// B fragment 32x16 f16 from "n-major" storage: element (k,n) at p0[n*ldn + k].
// lanes 0-15: N=lane, K in {0..7} U {16..23}; lanes 16-31: N=lane-16, K+8.
__device__ __forceinline__ v16h load_bfrag(const _Float16* p0, int ldn) {
  const int lane = threadIdx.x & 31;
  const int n  = lane & 15;
  const int kb = (lane >> 4) << 3;  // 0 or 8
  const _Float16* p = p0 + (size_t)n * ldn + kb;
  v16h b;
#pragma unroll
  for (int i = 0; i < 8; ++i) { b[i] = p[i]; b[8 + i] = p[16 + i]; }
  return b;
}

// ---------------- stage 1: precision conversion ----------------

__global__ void cvt_f16_kernel(const float* __restrict__ src,
                               _Float16* __restrict__ dst, int n) {
  int i = blockIdx.x * blockDim.x + threadIdx.x;
  if (i < n) dst[i] = (_Float16)src[i];
}

// W[k][n] (rows x cols) -> WT[n][k]
__global__ void cvt_transpose_kernel(const float* __restrict__ src,
                                     _Float16* __restrict__ dst,
                                     int rows, int cols) {
  int i = blockIdx.x * blockDim.x + threadIdx.x;
  if (i < rows * cols) {
    int k = i / cols;
    int n = i - k * cols;
    dst[(size_t)n * rows + k] = (_Float16)src[i];
  }
}

// ---------------- stage 2: QKV GEMM  (M=8192, N=3072, K=1024) ----------------
// 4-wave WG -> 128x64 tile; B tile (64x32) double-buffered in LDS via async
// copies overlapped with WMMA. Writes Q,K as [B,H,T,64], V^T as [B,H,64,T].

__global__ void __launch_bounds__(128)
qkv_wmma_kernel(const _Float16* __restrict__ xh, const _Float16* __restrict__ WaT,
                const float* __restrict__ bias,
                _Float16* __restrict__ Q, _Float16* __restrict__ Kt,
                _Float16* __restrict__ VT) {
  __shared__ _Float16 sB[2][64 * 32];  // [buf][n:64][k:32]
  const int tid = threadIdx.x;
  const int w = tid >> 5, lane = tid & 31;
  const int m0 = blockIdx.x * 128 + w * 32;
  const int n0 = blockIdx.y * 64;
  const int hi = lane >> 4, nl = lane & 15;
  const unsigned sbb = lds_off_of(&sB[0][0]);

  auto stageB = [&](int buf, int kk) {
    const int row = tid >> 1, off = (tid & 1) * 32;
    const char* g = (const char*)(WaT + (size_t)(n0 + row) * NEMBD + kk) + off;
    const unsigned l = sbb + (unsigned)buf * (64 * 32 * 2) + row * 64 + off;
    async_copy_b128(l, g);
    async_copy_b128(l + 16, g + 16);
  };

  v8f acc[2][4];
#pragma unroll
  for (int i = 0; i < 2; ++i)
#pragma unroll
    for (int j = 0; j < 4; ++j) acc[i][j] = zero8();

  stageB(0, 0);                        // prologue: tile 0 in flight
  for (int t = 0; t < NEMBD / 32; ++t) {
    const int cur = t & 1;
    const int kk = t * 32;
    __syncthreads();                   // buf[cur^1] consumers (iter t-1) done
    if (kk + 32 < NEMBD) {
      stageB(cur ^ 1, kk + 32);        // next tile streams behind compute
      __builtin_prefetch(xh + (size_t)m0 * NEMBD + kk + 32, 0, 3);
      wait_async_le2();                // oldest (cur) tile landed
    } else {
      wait_async_le0();
    }
    __syncthreads();                   // cur tile visible to all waves

    v16h a0 = load_afrag(xh + (size_t)(m0)      * NEMBD + kk, NEMBD);
    v16h a1 = load_afrag(xh + (size_t)(m0 + 16) * NEMBD + kk, NEMBD);
#pragma unroll
    for (int j = 0; j < 4; ++j) {
      v16h bf = load_bfrag(&sB[cur][16 * j * 32], 32);
      acc[0][j] = wmma16(a0, bf, acc[0][j]);
      acc[1][j] = wmma16(a1, bf, acc[1][j]);
    }
  }

  const int section = n0 >> 10;        // 0=Q, 1=K, 2=V (uniform per block)
  const int ninsec  = n0 & 1023;
#pragma unroll
  for (int i = 0; i < 2; ++i) {
#pragma unroll
    for (int j = 0; j < 4; ++j) {
      const int ncol = ninsec + 16 * j + nl;
      const int h = ncol >> 6, d = ncol & 63;
      const float bv = bias[n0 + 16 * j + nl];
#pragma unroll
      for (int r = 0; r < 8; ++r) {
        const int m = m0 + 16 * i + r + 8 * hi;
        const int b = m >> 11, t = m & (TSEQ - 1);
        const float v = acc[i][j][r] + bv;
        const size_t bh = (size_t)(b * NHEAD + h);
        if (section == 0)      Q [(bh * TSEQ + t) * HDIM + d] = (_Float16)v;
        else if (section == 1) Kt[(bh * TSEQ + t) * HDIM + d] = (_Float16)v;
        else                   VT[(bh * HDIM + d) * TSEQ + t] = (_Float16)v;
      }
    }
  }
}

// ---------------- stage 3: flash attention ----------------
// grid: (T/64, B*H); 4-wave WG owns 64 query rows; K/V tiles double-buffered
// in LDS via async copies, shared by all waves (4x less L2 traffic).

__global__ void __launch_bounds__(128)
fattn_wmma_kernel(const _Float16* __restrict__ Q, const _Float16* __restrict__ Kt,
                  const _Float16* __restrict__ VT, _Float16* __restrict__ Y) {
  __shared__ _Float16 sK[2][32 * 64];  // [buf][t:32][d:64]
  __shared__ _Float16 sV[2][64 * 32];  // [buf][d:64][t:32]  (V^T tile)
  __shared__ _Float16 sP[4][16 * 32];  // per-wave P relayout scratch
  const int tid = threadIdx.x;
  const int w = tid >> 5, lane = tid & 31;
  const int hi = lane >> 4, nl = lane & 15;
  const int bh = blockIdx.y;
  const int q0 = blockIdx.x * 64;
  const int qw = q0 + 16 * w;          // this wave's query-tile base

  const _Float16* Qb = Q  + (size_t)bh * TSEQ * HDIM;
  const _Float16* Kb = Kt + (size_t)bh * TSEQ * HDIM;
  const _Float16* Vb = VT + (size_t)bh * HDIM * TSEQ;
  const unsigned skb = lds_off_of(&sK[0][0]);
  const unsigned svb = lds_off_of(&sV[0][0]);

  auto stageKV = [&](int buf, int j) {
    {  // K tile: 4KB contiguous
      const char* g = (const char*)(Kb + (size_t)j * HDIM) + tid * 32;
      const unsigned l = skb + (unsigned)buf * (32 * 64 * 2) + tid * 32;
      async_copy_b128(l, g);
      async_copy_b128(l + 16, g + 16);
    }
    {  // V^T tile: 64 rows x 64B
      const int row = tid >> 1, off = (tid & 1) * 32;
      const char* g = (const char*)(Vb + (size_t)row * TSEQ + j) + off;
      const unsigned l = svb + (unsigned)buf * (64 * 32 * 2) + row * 64 + off;
      async_copy_b128(l, g);
      async_copy_b128(l + 16, g + 16);
    }
  };

  const v16h aq0 = load_afrag(Qb + (size_t)qw * HDIM,      HDIM);
  const v16h aq1 = load_afrag(Qb + (size_t)qw * HDIM + 32, HDIM);

  float mrow[8], lrow[8];
  v8f o[4];
#pragma unroll
  for (int r = 0; r < 8; ++r) { mrow[r] = -1e30f; lrow[r] = 0.0f; }
#pragma unroll
  for (int c = 0; c < 4; ++c) o[c] = zero8();

  const int ntiles = (q0 + 64) / 32;        // causal bound for whole WG
  stageKV(0, 0);                            // prologue
  for (int t = 0; t < ntiles; ++t) {
    const int cur = t & 1;
    const int j = t * 32;
    __syncthreads();                        // buf[cur^1] consumers done
    if (t + 1 < ntiles) {
      stageKV(cur ^ 1, j + 32);             // next K/V stream behind compute
      wait_async_le4();                     // oldest (cur) tiles landed
    } else {
      wait_async_le0();
    }
    __syncthreads();                        // cur tiles visible to all waves

    if (j < qw + 16) {                      // wave-uniform causal predicate
      // ---- S = Q * K^T for keys [j, j+32): 4 WMMAs from LDS ----
      v16h kb0 = load_bfrag(&sK[cur][0],              HDIM);
      v16h kb1 = load_bfrag(&sK[cur][32],             HDIM);
      v16h kb2 = load_bfrag(&sK[cur][16 * HDIM],      HDIM);
      v16h kb3 = load_bfrag(&sK[cur][16 * HDIM + 32], HDIM);
      v8f s0 = zero8(), s1 = zero8();
      s0 = wmma16(aq0, kb0, s0); s0 = wmma16(aq1, kb1, s0);
      s1 = wmma16(aq0, kb2, s1); s1 = wmma16(aq1, kb3, s1);

      // ---- online softmax (scale 1/sqrt(64), causal, 16-lane reductions) ----
      float p0[8], p1[8], alpha[8];
#pragma unroll
      for (int r = 0; r < 8; ++r) {
        const int qm = qw + r + 8 * hi;
        float v0 = (j + nl)      <= qm ? s0[r] * 0.125f : -1e30f;
        float v1 = (j + 16 + nl) <= qm ? s1[r] * 0.125f : -1e30f;
        float mx = fmaxf(v0, v1);
        mx = fmaxf(mx, __shfl_xor(mx, 1, 32));
        mx = fmaxf(mx, __shfl_xor(mx, 2, 32));
        mx = fmaxf(mx, __shfl_xor(mx, 4, 32));
        mx = fmaxf(mx, __shfl_xor(mx, 8, 32));
        const float mnew = fmaxf(mrow[r], mx);
        alpha[r] = __expf(mrow[r] - mnew);
        mrow[r] = mnew;
        p0[r] = __expf(v0 - mnew);
        p1[r] = __expf(v1 - mnew);
        float sum = p0[r] + p1[r];
        sum += __shfl_xor(sum, 1, 32);
        sum += __shfl_xor(sum, 2, 32);
        sum += __shfl_xor(sum, 4, 32);
        sum += __shfl_xor(sum, 8, 32);
        lrow[r] = lrow[r] * alpha[r] + sum;
      }
#pragma unroll
      for (int c = 0; c < 4; ++c)
#pragma unroll
        for (int r = 0; r < 8; ++r) o[c][r] *= alpha[r];

      // ---- P: C-frag -> A-frag via per-wave LDS scratch ----
#pragma unroll
      for (int r = 0; r < 8; ++r) {
        const int ml = r + 8 * hi;
        sP[w][ml * 32 + nl]      = (_Float16)p0[r];
        sP[w][ml * 32 + 16 + nl] = (_Float16)p1[r];
      }
      v16h pa;                              // wave-local RAW; LDS pipe in-order
      {
        const _Float16* p = &sP[w][nl * 32 + (hi << 3)];
#pragma unroll
        for (int i = 0; i < 8; ++i) { pa[i] = p[i]; pa[8 + i] = p[16 + i]; }
      }

      // ---- O += P * V from LDS V^T tile ----
#pragma unroll
      for (int c = 0; c < 4; ++c) {
        v16h vb = load_bfrag(&sV[cur][16 * c * 32], 32);
        o[c] = wmma16(pa, vb, o[c]);
      }
    }
  }

  // ---- normalize and write Y[b][t][h*64+d] (f16) ----
  const int b = bh >> 4, h = bh & 15;
#pragma unroll
  for (int c = 0; c < 4; ++c) {
#pragma unroll
    for (int r = 0; r < 8; ++r) {
      const int m = qw + r + 8 * hi;
      const int d = 16 * c + nl;
      const float v = o[c][r] / lrow[r];
      Y[((size_t)b * TSEQ + m) * NEMBD + h * HDIM + d] = (_Float16)v;
    }
  }
}

// ---------------- stage 4: output projection (M=8192, N=1024, K=1024) ----------------

__global__ void __launch_bounds__(128)
proj_wmma_kernel(const _Float16* __restrict__ Yh, const _Float16* __restrict__ WpT,
                 const float* __restrict__ bias, float* __restrict__ out) {
  __shared__ _Float16 sB[2][64 * 32];
  const int tid = threadIdx.x;
  const int w = tid >> 5, lane = tid & 31;
  const int m0 = blockIdx.x * 128 + w * 32;
  const int n0 = blockIdx.y * 64;
  const int hi = lane >> 4, nl = lane & 15;
  const unsigned sbb = lds_off_of(&sB[0][0]);

  auto stageB = [&](int buf, int kk) {
    const int row = tid >> 1, off = (tid & 1) * 32;
    const char* g = (const char*)(WpT + (size_t)(n0 + row) * NEMBD + kk) + off;
    const unsigned l = sbb + (unsigned)buf * (64 * 32 * 2) + row * 64 + off;
    async_copy_b128(l, g);
    async_copy_b128(l + 16, g + 16);
  };

  v8f acc[2][4];
#pragma unroll
  for (int i = 0; i < 2; ++i)
#pragma unroll
    for (int j = 0; j < 4; ++j) acc[i][j] = zero8();

  stageB(0, 0);
  for (int t = 0; t < NEMBD / 32; ++t) {
    const int cur = t & 1;
    const int kk = t * 32;
    __syncthreads();
    if (kk + 32 < NEMBD) {
      stageB(cur ^ 1, kk + 32);
      __builtin_prefetch(Yh + (size_t)m0 * NEMBD + kk + 32, 0, 3);
      wait_async_le2();
    } else {
      wait_async_le0();
    }
    __syncthreads();

    v16h a0 = load_afrag(Yh + (size_t)(m0)      * NEMBD + kk, NEMBD);
    v16h a1 = load_afrag(Yh + (size_t)(m0 + 16) * NEMBD + kk, NEMBD);
#pragma unroll
    for (int j = 0; j < 4; ++j) {
      v16h bf = load_bfrag(&sB[cur][16 * j * 32], 32);
      acc[0][j] = wmma16(a0, bf, acc[0][j]);
      acc[1][j] = wmma16(a1, bf, acc[1][j]);
    }
  }

#pragma unroll
  for (int i = 0; i < 2; ++i) {
#pragma unroll
    for (int j = 0; j < 4; ++j) {
      const int n = n0 + 16 * j + nl;
      const float bv = bias[n];
#pragma unroll
      for (int r = 0; r < 8; ++r) {
        const int m = m0 + 16 * i + r + 8 * hi;
        out[(size_t)m * NEMBD + n] = acc[i][j][r] + bv;
      }
    }
  }
}

// ---------------- host launch ----------------

extern "C" void kernel_launch(void* const* d_in, const int* in_sizes, int n_in,
                              void* d_out, int out_size, void* d_ws, size_t ws_size,
                              hipStream_t stream) {
  (void)in_sizes; (void)n_in; (void)out_size; (void)ws_size;
  const float* x      = (const float*)d_in[0];   // [4,2048,1024]
  const float* W_attn = (const float*)d_in[1];   // [1024,3072]
  const float* b_attn = (const float*)d_in[2];   // [3072]
  const float* W_proj = (const float*)d_in[3];   // [1024,1024]
  const float* b_proj = (const float*)d_in[4];   // [1024]
  float* out = (float*)d_out;                    // [4,2048,1024]

  const int nX  = BSZ * TSEQ * NEMBD;            // 8388608
  const int nWa = NEMBD * 3 * NEMBD;             // 3145728
  const int nWp = NEMBD * NEMBD;                 // 1048576

  char* ws = (char*)d_ws;
  size_t off = 0;
  _Float16* xh  = (_Float16*)(ws + off); off += (size_t)nX  * 2;
  _Float16* WaT = (_Float16*)(ws + off); off += (size_t)nWa * 2;
  _Float16* WpT = (_Float16*)(ws + off); off += (size_t)nWp * 2;
  _Float16* Qd  = (_Float16*)(ws + off); off += (size_t)nX  * 2;
  _Float16* Kd  = (_Float16*)(ws + off); off += (size_t)nX  * 2;
  _Float16* VTd = (_Float16*)(ws + off); off += (size_t)nX  * 2;
  _Float16* Yh  = (_Float16*)(ws + off); off += (size_t)nX  * 2;

  // stage 1: conversions
  cvt_f16_kernel<<<(nX + 255) / 256, 256, 0, stream>>>(x, xh, nX);
  cvt_transpose_kernel<<<(nWa + 255) / 256, 256, 0, stream>>>(W_attn, WaT, NEMBD, 3 * NEMBD);
  cvt_transpose_kernel<<<(nWp + 255) / 256, 256, 0, stream>>>(W_proj, WpT, NEMBD, NEMBD);

  // stage 2: QKV gemm (WG = 128x64 tile, 4 waves)
  qkv_wmma_kernel<<<dim3((BSZ * TSEQ) / 128, (3 * NEMBD) / 64), 128, 0, stream>>>(
      xh, WaT, b_attn, Qd, Kd, VTd);

  // stage 3: flash attention (T/64 q-tiles x B*H, 4 waves each)
  fattn_wmma_kernel<<<dim3(TSEQ / 64, BSZ * NHEAD), 128, 0, stream>>>(Qd, Kd, VTd, Yh);

  // stage 4: projection
  proj_wmma_kernel<<<dim3((BSZ * TSEQ) / 128, NEMBD / 64), 128, 0, stream>>>(
      Yh, WpT, b_proj, out);
}
